// HSTU_40991167873566
// MI455X (gfx1250) — compile-verified, matrix operand order checked by reading, not run on hardware
//
#include <hip/hip_runtime.h>

// ---------------- constants (match reference) ----------------
#define BB    8
#define LL    1023
#define SS    1024
#define DD    512
#define NHH   8
#define HDD   64
#define NLL   4
#define MSLL  2048
#define D4    2048
#define MROWS (BB * SS)   // 8192

typedef __attribute__((ext_vector_type(8)))  __bf16 v8bf;
typedef __attribute__((ext_vector_type(16))) __bf16 v16bf;
typedef __attribute__((ext_vector_type(8)))  float  v8f;

// ---------------- WMMA fragment load ----------------
// 16x32 bf16 A-fragment (also used for B via transposed layouts):
// lane L (0..31): row r = L&15, K-base = (L&16)?8:0
// element e in 0..7  -> K = base + e        (contiguous)
// element e in 8..15 -> K = base + 8 + e    (contiguous, starts at base+16)
__device__ __forceinline__ v16bf load_frag(const __bf16* p, int ld, int lane) {
    int r  = lane & 15;
    int kb = (lane & 16) ? 8 : 0;
    const __bf16* q = p + (size_t)r * ld + kb;
    v8bf lo = *(const v8bf*)(q);
    v8bf hi = *(const v8bf*)(q + 16);
    v16bf out;
#pragma unroll
    for (int i = 0; i < 8; ++i) { out[i] = lo[i]; out[i + 8] = hi[i]; }
    return out;
}

// ---------------- CDNA5 async global->LDS copy (ASYNCcnt-tracked) ----------------
// per-lane: copy 16B from global addr to LDS byte offset (ISA 15.18.3 op 98)
__device__ __forceinline__ void async_cp16(unsigned lds_off, const __bf16* gptr) {
    asm volatile("global_load_async_to_lds_b128 %0, %1, off"
                 :: "v"(lds_off), "v"((unsigned long long)(size_t)gptr)
                 : "memory");
}
__device__ __forceinline__ void async_wait0() {
    asm volatile("s_wait_asynccnt 0" ::: "memory");
}
__device__ __forceinline__ unsigned lds_off_of(const void* p) {
    return (unsigned)(size_t)p;   // generic addr low 32 bits = LDS byte offset
}

// fast silu: one v_exp + one v_rcp (avoids IEEE div expansion: ~10 VALU -> ~4 VALU)
__device__ __forceinline__ float silu_f(float x) {
    return x * __builtin_amdgcn_rcpf(1.0f + __expf(-x));
}

// ---------------- embedding build ----------------
__global__ void embed_kernel(const int* __restrict__ hist_ids, const int* __restrict__ hist_cids,
                             const int* __restrict__ hist_len,
                             const int* __restrict__ tgt_id, const int* __restrict__ tgt_cid,
                             const float* __restrict__ item_emb, const float* __restrict__ cate_emb,
                             const float* __restrict__ seg_emb,
                             float* __restrict__ h, __bf16* __restrict__ hbf) {
    int bs = blockIdx.x;           // b*S + s
    int b = bs >> 10, s = bs & (SS - 1);
    int len = hist_len[b];
    int id, cid, sg;
    if (s == len)      { id = tgt_id[b]; cid = tgt_cid[b]; sg = 1; }
    else if (s < LL)   { id = hist_ids[b * LL + s]; cid = hist_cids[b * LL + s]; sg = 0; }
    else               { id = 0; cid = 0; sg = 0; }
    const float* pe = item_emb + (size_t)id * DD;
    const float* pc = cate_emb + (size_t)cid * DD;
    const float* ps = seg_emb + (size_t)sg * DD;
    for (int c = threadIdx.x; c < DD; c += blockDim.x) {
        float v = pe[c] + pc[c] + ps[c];
        h[(size_t)bs * DD + c]   = v;
        hbf[(size_t)bs * DD + c] = (__bf16)v;
    }
}

// ---------------- weight transpose + bf16 convert: W[NL,K,N] -> Wt[NL,N,K] ----------------
__global__ void wt_transpose(const float* __restrict__ W, __bf16* __restrict__ Wt, int K, int N) {
    size_t idx = (size_t)blockIdx.x * blockDim.x + threadIdx.x;
    size_t per = (size_t)K * N;
    size_t total = (size_t)NLL * per;
    if (idx >= total) return;
    size_t l = idx / per, rem = idx % per;
    int k = (int)(rem / N), n = (int)(rem % N);
    Wt[l * per + (size_t)n * K + k] = (__bf16)W[idx];
}

// ---------------- GEMM: C[M,N] = act(A[M,K] @ Bt[N,K]^T + bias[N]) ----------------
// 256 threads = 8 waves (2 m x 4 n); block tile 128x256; wave tile 64x64 (4x4 WMMA accums)
// A/B k-slices staged to LDS with async copies, double-buffered.
template <bool SILU>
__global__ void gemm_bf16(const __bf16* __restrict__ A, const __bf16* __restrict__ Bt,
                          const float* __restrict__ bias, float* __restrict__ C,
                          int M, int N, int K) {
    __shared__ __bf16 a_tile[2][128][32];   // 16 KB
    __shared__ __bf16 b_tile[2][256][32];   // 32 KB
    int tid = threadIdx.x;
    int lane = tid & 31, wave = tid >> 5;
    int wm = wave >> 2, wn = wave & 3;
    int mblk = blockIdx.x * 128, nblk = blockIdx.y * 256;

    auto stage = [&](int buf, int k0) {
        int kc  = (tid & 3) * 8;
#pragma unroll
        for (int it = 0; it < 2; ++it) {
            int row = (tid >> 2) + it * 64;
            async_cp16(lds_off_of(&a_tile[buf][row][kc]),
                       A + (size_t)(mblk + row) * K + (k0 + kc));
        }
#pragma unroll
        for (int it = 0; it < 4; ++it) {
            int row = (tid >> 2) + it * 64;
            async_cp16(lds_off_of(&b_tile[buf][row][kc]),
                       Bt + (size_t)(nblk + row) * K + (k0 + kc));
        }
    };

    v8f acc[4][4] = {};
    const int KT = K / 32;
    stage(0, 0);
    async_wait0();
    __syncthreads();
    for (int kt = 0; kt < KT; ++kt) {
        int buf = kt & 1;
        if (kt + 1 < KT) stage(buf ^ 1, (kt + 1) * 32);   // async prefetch next slice
        v16bf a[4], bw[4];
#pragma unroll
        for (int i = 0; i < 4; ++i)
            a[i] = load_frag(&a_tile[buf][wm * 64 + 16 * i][0], 32, lane);
#pragma unroll
        for (int j = 0; j < 4; ++j)
            bw[j] = load_frag(&b_tile[buf][wn * 64 + 16 * j][0], 32, lane);
#pragma unroll
        for (int i = 0; i < 4; ++i)
#pragma unroll
            for (int j = 0; j < 4; ++j)
                acc[i][j] = __builtin_amdgcn_wmma_f32_16x16x32_bf16(
                    false, a[i], false, bw[j], (short)0, acc[i][j], false, false);
        if (kt + 1 < KT) { async_wait0(); __syncthreads(); }
    }

    int lrow = lane & 15, lhi = (lane >> 4) * 8;
#pragma unroll
    for (int i = 0; i < 4; ++i)
#pragma unroll
        for (int j = 0; j < 4; ++j)
#pragma unroll
            for (int r = 0; r < 8; ++r) {
                int m = mblk + wm * 64 + 16 * i + lhi + r;
                int n = nblk + wn * 64 + 16 * j + lrow;
                float x = acc[i][j][r] + bias[n];
                if (SILU) x = silu_f(x);
                C[(size_t)m * N + n] = x;
            }
}

// ---------------- pack q,k,v from z ----------------
// z[B,S,4D]: u=[0:512] q=[512:1024] k=[1024:1536] v=[1536:2048]
// qb,kb: [B*NH, S, 64] bf16 ; vtb: [B*NH, 64, S] bf16 (transposed for contiguous-K B frags)
__global__ void pack_qkv(const float* __restrict__ z, __bf16* __restrict__ qb,
                         __bf16* __restrict__ kb, __bf16* __restrict__ vtb) {
    int idx = blockIdx.x * blockDim.x + threadIdx.x;   // B*NH*S*64 = 4M
    int d = idx & 63;
    int s = (idx >> 6) & (SS - 1);
    int bh = idx >> 16;
    int b = bh >> 3, hh = bh & 7;
    size_t zrow = ((size_t)(b * SS + s)) * D4 + hh * 64 + d;
    qb[((size_t)bh * SS + s) * 64 + d]  = (__bf16)z[zrow + 512];
    kb[((size_t)bh * SS + s) * 64 + d]  = (__bf16)z[zrow + 1024];
    vtb[((size_t)bh * 64 + d) * SS + s] = (__bf16)z[zrow + 1536];
}

// ---------------- attention: y[b,s,h*64+d] = (silu(qk^T + bias) * causal) @ v * u ----------------
// grid.x = B*NH, grid.y = 8 (s blocks of 128); 8 waves, each wave owns 16 query rows.
// K/V tiles for each 32-step of t are async-staged to LDS, shared by all 8 waves, double-buffered.
// Relative-position bias window is staged to LDS once per block (bias reads -> ds, not vmem).
__global__ void attn_kernel(const __bf16* __restrict__ qb, const __bf16* __restrict__ kb,
                            const __bf16* __restrict__ vtb, const float* __restrict__ z,
                            const float* __restrict__ posw, float* __restrict__ yout) {
    __shared__ __bf16 k_tile[2][32][64];    // 8 KB  (rows = t-offset)
    __shared__ __bf16 vt_tile[2][64][32];   // 8 KB  (rows = d)
    __shared__ __bf16 att_lds[8][16][32];   // 8 KB  (per-wave score relayout)
    __shared__ float  pos_lds[1040];        // 4.1 KB bias window for this block

    int bh = blockIdx.x;
    int b = bh >> 3, hh = bh & 7;
    int tid = threadIdx.x, lane = tid & 31, wave = tid >> 5;
    int sb = blockIdx.y * 128;
    int s_base = sb + wave * 16;
    int lrow = lane & 15, lhi = (lane >> 4) * 8;

    const __bf16* qp = qb + ((size_t)bh * SS + s_base) * 64;
    v16bf a0 = load_frag(qp,      64, lane);   // d = 0..31
    v16bf a1 = load_frag(qp + 32, 64, lane);   // d = 32..63

    auto stage = [&](int buf, int t0) {
        {   // K: 32 rows x 64 d  (32 rows x 8 chunks of 8 bf16)
            int row = tid >> 3, kc = (tid & 7) * 8;
            async_cp16(lds_off_of(&k_tile[buf][row][kc]),
                       kb + ((size_t)bh * SS + t0 + row) * 64 + kc);
        }
        {   // V^T: 64 rows(d) x 32 t  (64 rows x 4 chunks)
            int row = tid >> 2, kc = (tid & 3) * 8;
            async_cp16(lds_off_of(&vt_tile[buf][row][kc]),
                       vtb + ((size_t)bh * 64 + row) * SS + t0 + kc);
        }
    };

    v8f yacc[4] = {};
    int t_end = s_base + 16;          // exclusive causal bound for this wave
    int nT = (sb + 128) / 32;         // uniform block-wide t-tile count
    stage(0, 0);
    // stage bias window: global idx W0 = MSL-1-(sb+127) = 1920-sb, lds idx = g - W0
    for (int i = tid; i < sb + 144; i += 256)
        pos_lds[i] = posw[(1920 - sb) + i];
    async_wait0();
    __syncthreads();
    for (int tt = 0; tt < nT; ++tt) {
        int t0 = tt * 32, buf = tt & 1;
        if (tt + 1 < nT) stage(buf ^ 1, t0 + 32);
        if (t0 < t_end) {
#pragma unroll
            for (int half = 0; half < 2; ++half) {
                int th = t0 + half * 16;
                if (th < t_end) {
                    v16bf b0 = load_frag(&k_tile[buf][half * 16][0],  64, lane);
                    v16bf b1 = load_frag(&k_tile[buf][half * 16][32], 64, lane);
                    v8f c = {};
                    c = __builtin_amdgcn_wmma_f32_16x16x32_bf16(false, a0, false, b0, (short)0, c, false, false);
                    c = __builtin_amdgcn_wmma_f32_16x16x32_bf16(false, a1, false, b1, (short)0, c, false, false);
                    // lds bias index: (MSL-1 + tcol - srow) - (1920 - sb) = tcol - srow + sb + 127
                    int tcol = th + lrow;
                    int bias_base = tcol - (s_base + lhi) + sb + 127;  // index for r=0
#pragma unroll
                    for (int r = 0; r < 8; ++r) {
                        int srow = s_base + lhi + r;
                        float x = c[r] + pos_lds[bias_base - r];
                        x = silu_f(x);
                        if (tcol > srow) x = 0.0f;
                        att_lds[wave][lhi + r][half * 16 + lrow] = (__bf16)x;
                    }
                } else {
#pragma unroll
                    for (int r = 0; r < 8; ++r)
                        att_lds[wave][lhi + r][half * 16 + lrow] = (__bf16)0.0f;
                }
            }
            // re-layout scores (16x32) -> A fragment via LDS, then y += att @ V
            v16bf aA = load_frag(&att_lds[wave][0][0], 32, lane);
#pragma unroll
            for (int j = 0; j < 4; ++j) {
                v16bf vB = load_frag(&vt_tile[buf][j * 16][0], 32, lane);
                yacc[j] = __builtin_amdgcn_wmma_f32_16x16x32_bf16(
                    false, aA, false, vB, (short)0, yacc[j], false, false);
            }
        }
        if (tt + 1 < nT) { async_wait0(); __syncthreads(); }
    }

    // store y * u into full [B,S,D] layout
#pragma unroll
    for (int j = 0; j < 4; ++j)
#pragma unroll
        for (int r = 0; r < 8; ++r) {
            int srow = s_base + lhi + r;
            int dcol = j * 16 + lrow;
            size_t row = (size_t)(b * SS + srow);
            float u = z[row * D4 + hh * 64 + dcol];          // u slice (offset 0)
            yout[row * DD + hh * 64 + dcol] = yacc[j][r] * u;
        }
}

// ---------------- layernorm (optional residual, optional fp32/bf16 outputs) ----------------
__global__ void layernorm_kernel(const float* __restrict__ x, const float* __restrict__ res,
                                 const float* __restrict__ w, const float* __restrict__ bp,
                                 float* __restrict__ outf, __bf16* __restrict__ outb) {
    int row = blockIdx.x;
    const float* xp = x + (size_t)row * DD;
    float v[2], s = 0.f, s2 = 0.f;
#pragma unroll
    for (int i = 0; i < 2; ++i) {
        int c = threadIdx.x + i * 256;
        float t = xp[c];
        if (res) t += res[(size_t)row * DD + c];
        v[i] = t; s += t; s2 += t * t;
    }
    __shared__ float r1[256], r2[256];
    r1[threadIdx.x] = s; r2[threadIdx.x] = s2;
    __syncthreads();
    for (int off = 128; off > 0; off >>= 1) {
        if (threadIdx.x < off) {
            r1[threadIdx.x] += r1[threadIdx.x + off];
            r2[threadIdx.x] += r2[threadIdx.x + off];
        }
        __syncthreads();
    }
    float mu  = r1[0] * (1.0f / DD);
    float var = r2[0] * (1.0f / DD) - mu * mu;
    float inv = rsqrtf(var + 1e-5f);
#pragma unroll
    for (int i = 0; i < 2; ++i) {
        int c = threadIdx.x + i * 256;
        float o = (v[i] - mu) * inv * w[c] + bp[c];
        if (outf) outf[(size_t)row * DD + c] = o;
        if (outb) outb[(size_t)row * DD + c] = (__bf16)o;
    }
}

// ---------------- host launch ----------------
extern "C" void kernel_launch(void* const* d_in, const int* in_sizes, int n_in,
                              void* d_out, int out_size, void* d_ws, size_t ws_size,
                              hipStream_t stream) {
    const int*   hist_ids  = (const int*)d_in[0];
    const int*   hist_cids = (const int*)d_in[1];
    const int*   hist_len  = (const int*)d_in[2];
    const int*   tgt_id    = (const int*)d_in[3];
    const int*   tgt_cid   = (const int*)d_in[4];
    const float* item_emb  = (const float*)d_in[5];
    const float* cate_emb  = (const float*)d_in[6];
    const float* seg_emb   = (const float*)d_in[7];
    const float* W1        = (const float*)d_in[8];
    const float* b1        = (const float*)d_in[9];
    const float* W2        = (const float*)d_in[10];
    const float* b2        = (const float*)d_in[11];
    const float* ln1_w     = (const float*)d_in[12];
    const float* ln1_b     = (const float*)d_in[13];
    const float* ln2_w     = (const float*)d_in[14];
    const float* ln2_b     = (const float*)d_in[15];
    const float* pos_w     = (const float*)d_in[16];
    const float* lnf_w     = (const float*)d_in[17];
    const float* lnf_b     = (const float*)d_in[18];
    float* out = (float*)d_out;

    // workspace carve-out
    char* ws = (char*)d_ws;
    auto alloc = [&](size_t bytes) {
        void* p = ws;
        ws += (bytes + 255) & ~(size_t)255;
        return p;
    };
    float*  h     = (float*)alloc((size_t)MROWS * DD * 4);           // 16 MB
    __bf16* hbf   = (__bf16*)alloc((size_t)MROWS * DD * 2);          //  8 MB
    float*  zbuf  = (float*)alloc((size_t)MROWS * D4 * 4);           // 64 MB
    __bf16* qbuf  = (__bf16*)alloc((size_t)MROWS * DD * 2);          //  8 MB
    __bf16* kbuf  = (__bf16*)alloc((size_t)MROWS * DD * 2);          //  8 MB
    __bf16* vtbuf = (__bf16*)alloc((size_t)MROWS * DD * 2);          //  8 MB
    float*  ybuf  = (float*)alloc((size_t)MROWS * DD * 4);           // 16 MB
    __bf16* ln1bf = (__bf16*)alloc((size_t)MROWS * DD * 2);          //  8 MB
    float*  y2    = (float*)alloc((size_t)MROWS * DD * 4);           // 16 MB
    __bf16* W1t   = (__bf16*)alloc((size_t)NLL * DD * D4 * 2);       //  8 MB
    __bf16* W2t   = (__bf16*)alloc((size_t)NLL * DD * DD * 2);       //  2 MB

    // weight conversion (bf16, transposed to [N,K])
    {
        size_t t1 = (size_t)NLL * DD * D4;
        wt_transpose<<<dim3((unsigned)((t1 + 255) / 256)), 256, 0, stream>>>(W1, W1t, DD, D4);
        size_t t2 = (size_t)NLL * DD * DD;
        wt_transpose<<<dim3((unsigned)((t2 + 255) / 256)), 256, 0, stream>>>(W2, W2t, DD, DD);
    }

    embed_kernel<<<dim3(MROWS), 256, 0, stream>>>(hist_ids, hist_cids, hist_len, tgt_id, tgt_cid,
                                                  item_emb, cate_emb, seg_emb, h, hbf);

    for (int l = 0; l < NLL; ++l) {
        // z = silu(h @ W1 + b1)   [8192 x 2048]
        gemm_bf16<true><<<dim3(MROWS / 128, D4 / 256), 256, 0, stream>>>(
            hbf, W1t + (size_t)l * DD * D4, b1 + (size_t)l * D4, zbuf, MROWS, D4, DD);

        pack_qkv<<<dim3((BB * NHH * SS * HDD) / 256), 256, 0, stream>>>(zbuf, qbuf, kbuf, vtbuf);

        attn_kernel<<<dim3(BB * NHH, SS / 128), 256, 0, stream>>>(
            qbuf, kbuf, vtbuf, zbuf, pos_w + (size_t)l * (2 * MSLL - 1), ybuf);

        // ln1 (bf16 out only, feeds GEMM2)
        layernorm_kernel<<<dim3(MROWS), 256, 0, stream>>>(
            ybuf, nullptr, ln1_w + (size_t)l * DD, ln1_b + (size_t)l * DD, nullptr, ln1bf);

        // y2 = ln1out @ W2 + b2   [8192 x 512]
        gemm_bf16<false><<<dim3(MROWS / 128, DD / 256), 256, 0, stream>>>(
            ln1bf, W2t + (size_t)l * DD * DD, b2 + (size_t)l * DD, y2, MROWS, DD, DD);

        // h = LN(y2 + h) ; also refresh bf16 copy for next layer GEMM1
        layernorm_kernel<<<dim3(MROWS), 256, 0, stream>>>(
            y2, h, ln2_w + (size_t)l * DD, ln2_b + (size_t)l * DD, h, hbf);
    }

    // final LN -> output fp32 [B,S,D]
    layernorm_kernel<<<dim3(MROWS), 256, 0, stream>>>(h, nullptr, lnf_w, lnf_b, out, nullptr);
}
